// HMM_74010876445131
// MI455X (gfx1250) — matrix operand order repaired and checked by reading, not compiled, not run on hardware
//
#include <hip/hip_runtime.h>

// ---------------------------------------------------------------------------
// Problem constants (from reference): N=64, T=256, K=64, V=32000, E=100
// ---------------------------------------------------------------------------
#define HN   64
#define HT   256
#define HK   64
#define HV   32000
#define HE   100
#define HEP  128                 // E padded to multiple of 32 for bf16 WMMA
#define HTM1 255                 // T-1
#define HNT  (HN * HTM1)         // 16320 (n,t) rows
#define HC   (HK * HK)           // 4096 transition columns

typedef __bf16 bf16x16 __attribute__((ext_vector_type(16)));
typedef float  f32x8   __attribute__((ext_vector_type(8)));

union Frag32B { bf16x16 v; uint4 u[2]; };

__device__ __forceinline__ unsigned short f32_to_bf16(float f) {
    unsigned u = __float_as_uint(f);
    u += 0x7FFFu + ((u >> 16) & 1u);          // round-to-nearest-even
    return (unsigned short)(u >> 16);
}
__device__ __forceinline__ float bf16_to_f32(unsigned short h) {
    return __uint_as_float(((unsigned)h) << 16);
}

// ---------------------------------------------------------------------------
// Prep 1: W_trans (4096 x 100 f32) -> bf16 in WMMA B-fragment order.
// Layout: wfrag[((ct*4 + kk)*32 + lane)*16 + i]
//   column c = ct*16 + (lane&15), e = kk*32 + (lane>>4)*16 + i  (0 if e>=100)
// ---------------------------------------------------------------------------
__global__ void hmm_prep_w(const float* __restrict__ W, unsigned short* __restrict__ wfrag) {
    int idx = blockIdx.x * 256 + threadIdx.x;          // < 4096*128 = 524288
    int i    = idx & 15;
    int lane = (idx >> 4) & 31;
    int kk   = (idx >> 9) & 3;
    int ct   = idx >> 11;
    int c = ct * 16 + (lane & 15);
    int e = kk * 32 + ((lane >> 4) << 4) + i;
    float v = (e < HE) ? W[(size_t)c * HE + e] : 0.0f;
    wfrag[idx] = f32_to_bf16(v);
}

// ---------------------------------------------------------------------------
// Prep 2: gather xe = emb_weight[x[:, :-1]] -> bf16, rows padded to 128
// ---------------------------------------------------------------------------
__global__ void hmm_prep_xe(const int* __restrict__ x, const float* __restrict__ emb,
                            unsigned short* __restrict__ xeb) {
    int idx = blockIdx.x * 256 + threadIdx.x;          // < 16320*128
    int e = idx & (HEP - 1);
    int r = idx >> 7;
    int n = r / HTM1, t = r % HTM1;
    int tok = x[n * HT + t];
    float v = (e < HE) ? emb[(size_t)tok * HE + e] : 0.0f;
    xeb[idx] = f32_to_bf16(v);
}

// ---------------------------------------------------------------------------
// Main GEMM: scores[r, c] = xe[r,:] . W_trans[c,:] with fused 64-column-group
// log-softmax, bf16 output. Block = 512 thr (16 waves); block covers 16 rows
// x 4096 cols. Each wave owns 16 contiguous 16x16 C tiles = 4 i-groups, and
// processes them GROUP BY GROUP: only 4 accumulators (32 VGPRs) + 4 cached A
// fragments (32 VGPRs) are live at once -> no scratch spills, and the
// exp/log/shfl epilogue of group g co-executes with group g+1's WMMAs.
// ---------------------------------------------------------------------------
__global__ void __launch_bounds__(512)
hmm_gemm_tran(const unsigned short* __restrict__ xeb,
              const unsigned short* __restrict__ wfrag,
              unsigned short* __restrict__ tran) {
    const int tid  = threadIdx.x;
    const int lane = tid & 31;
    const int wave = tid >> 5;
    const int half = lane >> 4;
    const int m16  = lane & 15;
    const int rbase = blockIdx.x * 16;

    // A fragments for all 4 K-steps, loaded once, reused by all 4 groups.
    const unsigned short* arow = xeb + (size_t)(rbase + m16) * HEP + half * 16;
    Frag32B a[4];
#pragma unroll
    for (int kk = 0; kk < 4; ++kk) {
        a[kk].u[0] = *(const uint4*)(arow + kk * 32);        // K = kk*32 + half*16 ..
        a[kk].u[1] = *(const uint4*)(arow + kk * 32 + 8);    // K = kk*32 + half*16 + 8 ..
    }

#pragma unroll
    for (int g = 0; g < 4; ++g) {
        f32x8 acc[4];
#pragma unroll
        for (int tt = 0; tt < 4; ++tt)
#pragma unroll
            for (int rr = 0; rr < 8; ++rr) acc[tt][rr] = 0.0f;

#pragma unroll
        for (int kk = 0; kk < 4; ++kk) {
#pragma unroll
            for (int tt = 0; tt < 4; ++tt) {
                const int ctg = wave * 16 + g * 4 + tt;
                const uint4* bp =
                    (const uint4*)(wfrag + (((size_t)ctg * 4 + kk) * 32 + lane) * 16);
                Frag32B b;
                b.u[0] = bp[0];
                b.u[1] = bp[1];
                acc[tt] = __builtin_amdgcn_wmma_f32_16x16x32_bf16(
                    false, a[kk].v, false, b.v, (short)0, acc[tt], false, false);
            }
        }

        // Epilogue for this group: per row (M), LSE across 16 lanes of the
        // half-wave x 4 tiles (= one 64-column i-group), write bf16.
#pragma unroll
        for (int rr = 0; rr < 8; ++rr) {
            const int row = rbase + rr + (half << 3);
            float v0 = acc[0][rr];
            float v1 = acc[1][rr];
            float v2 = acc[2][rr];
            float v3 = acc[3][rr];
            float mx = fmaxf(fmaxf(v0, v1), fmaxf(v2, v3));
#pragma unroll
            for (int msk = 1; msk <= 8; msk <<= 1)
                mx = fmaxf(mx, __shfl_xor(mx, msk, 32));
            float s = __expf(v0 - mx) + __expf(v1 - mx) + __expf(v2 - mx) + __expf(v3 - mx);
#pragma unroll
            for (int msk = 1; msk <= 8; msk <<= 1)
                s += __shfl_xor(s, msk, 32);
            float lse = mx + __logf(s);
            size_t base = (size_t)row * HC + (size_t)(wave * 16 + 4 * g) * 16 + m16;
            tran[base]      = f32_to_bf16(v0 - lse);
            tran[base + 16] = f32_to_bf16(v1 - lse);
            tran[base + 32] = f32_to_bf16(v2 - lse);
            tran[base + 48] = f32_to_bf16(v3 - lse);
        }
    }
}

// ---------------------------------------------------------------------------
// Elog stage A: per-(vocab-block, k) partial max / sumexp of
// score[k, v] = emb_cluster_w[k,:] . vocab_w[v,:]
// Grid 125 x 256 threads (one v per thread).
// ---------------------------------------------------------------------------
__global__ void __launch_bounds__(256)
hmm_elog_partial(const float* __restrict__ ecw, const float* __restrict__ vw,
                 float* __restrict__ pmax, float* __restrict__ psum) {
    __shared__ float ecw_s[HK * HK];       // 16 KB
    __shared__ float sc[256 * 65];         // padded to kill bank conflicts, 65 KB
    const int tid = threadIdx.x;
    for (int i = tid; i < HK * HK; i += 256) ecw_s[i] = ecw[i];

    const int v = blockIdx.x * 256 + tid;  // 125*256 = 32000 exactly
    float vr[64];
    const float4* vp = (const float4*)(vw + (size_t)v * HK);
#pragma unroll
    for (int q = 0; q < 16; ++q) {
        float4 f = vp[q];
        vr[4 * q + 0] = f.x; vr[4 * q + 1] = f.y;
        vr[4 * q + 2] = f.z; vr[4 * q + 3] = f.w;
    }
    __syncthreads();

    for (int k = 0; k < HK; ++k) {
        const float* er = ecw_s + k * HK;
        float d = 0.0f;
#pragma unroll
        for (int e = 0; e < HK; ++e) d += er[e] * vr[e];
        sc[tid * 65 + k] = d;
    }
    __syncthreads();

    if (tid < HK) {
        float m = -1e30f;
        for (int t2 = 0; t2 < 256; ++t2) m = fmaxf(m, sc[t2 * 65 + tid]);
        float s = 0.0f;
        for (int t2 = 0; t2 < 256; ++t2) s += __expf(sc[t2 * 65 + tid] - m);
        pmax[blockIdx.x * HK + tid] = m;
        psum[blockIdx.x * HK + tid] = s;
    }
}

// ---------------------------------------------------------------------------
// Elog stage B: combine 125 partials -> elog_lse[k]; also alpha0 = log_softmax
// of start_w + start_b. One block, 64 threads. Fixed-order => deterministic.
// ---------------------------------------------------------------------------
__global__ void hmm_elog_final(const float* __restrict__ pmax, const float* __restrict__ psum,
                               const float* __restrict__ start_w, const float* __restrict__ start_b,
                               float* __restrict__ elog_lse, float* __restrict__ alpha0) {
    const int k = threadIdx.x;   // 64
    float m = -1e30f;
    for (int b = 0; b < 125; ++b) m = fmaxf(m, pmax[b * HK + k]);
    float s = 0.0f;
    for (int b = 0; b < 125; ++b) s += psum[b * HK + k] * __expf(pmax[b * HK + k] - m);
    elog_lse[k] = m + __logf(s);

    float a = start_w[k] + start_b[k];
    __shared__ float red[HK];
    __shared__ float lsev;
    red[k] = a;
    __syncthreads();
    if (k == 0) {
        float M = red[0];
        for (int i = 1; i < HK; ++i) M = fmaxf(M, red[i]);
        float S = 0.0f;
        for (int i = 0; i < HK; ++i) S += __expf(red[i] - M);
        lsev = M + __logf(S);
    }
    __syncthreads();
    alpha0[k] = a - lsev;
}

// ---------------------------------------------------------------------------
// Emit gather: emit[r, k] = ecw[k,:] . vocab_w[x[n,t+1],:] - elog_lse[k]
// Grid = 16320 blocks x 64 threads.
// ---------------------------------------------------------------------------
__global__ void hmm_emit_gather(const int* __restrict__ x, const float* __restrict__ ecw,
                                const float* __restrict__ vw, const float* __restrict__ elog_lse,
                                float* __restrict__ emit) {
    const int r = blockIdx.x;
    const int k = threadIdx.x;        // 64
    const int n = r / HTM1, t = r % HTM1;
    const int tok = x[n * HT + t + 1];
    __shared__ float vrow[HK];
    vrow[k] = vw[(size_t)tok * HK + k];
    __syncthreads();
    const float4* ep = (const float4*)(ecw + k * HK);
    float d = 0.0f;
#pragma unroll
    for (int q = 0; q < 16; ++q) {
        float4 f = ep[q];
        d += f.x * vrow[4 * q + 0] + f.y * vrow[4 * q + 1] +
             f.z * vrow[4 * q + 2] + f.w * vrow[4 * q + 3];
    }
    emit[(size_t)r * HK + k] = d - elog_lse[k];
}

// ---------------------------------------------------------------------------
// Forward scan: 64 blocks (one per sequence), 256 threads.
// Thread (j = tid&63, seg = tid>>6) owns a 16-i LSE segment of column j.
// tran tiles (8 KB / step) are prefetched to registers and double-buffered
// through LDS so global latency overlaps the dependent-chain compute.
// ---------------------------------------------------------------------------
__global__ void __launch_bounds__(256)
hmm_scan(const unsigned short* __restrict__ tran, const float* __restrict__ emit,
         const float* __restrict__ alpha0, float* __restrict__ nlse) {
    const int n   = blockIdx.x;
    const int tid = threadIdx.x;
    const int j   = tid & 63;
    const int seg = tid >> 6;

    __shared__ float alpha[HK];
    __shared__ float pm[4][HK];
    __shared__ float ps[4][HK];
    __shared__ unsigned short trbuf[2][HC];   // 16 KB double buffer

    if (tid < HK) alpha[tid] = alpha0[tid];

    const size_t r0 = (size_t)n * HTM1;
    const uint4* src0 = (const uint4*)(tran + r0 * HC);
    uint4 pa = src0[tid * 2], pb = src0[tid * 2 + 1];
    __syncthreads();

    for (int t = 0; t < HTM1; ++t) {
        const int buf = t & 1;
        ((uint4*)trbuf[buf])[tid * 2]     = pa;
        ((uint4*)trbuf[buf])[tid * 2 + 1] = pb;
        __syncthreads();

        if (t < HTM1 - 1) {   // prefetch next step while this step computes
            const uint4* s2 = (const uint4*)(tran + (r0 + t + 1) * HC);
            pa = s2[tid * 2];
            pb = s2[tid * 2 + 1];
        }

        float vv[16];
#pragma unroll
        for (int ii = 0; ii < 16; ++ii) {
            const int i = seg * 16 + ii;
            vv[ii] = alpha[i] + bf16_to_f32(trbuf[buf][i * HK + j]);
        }
        float m = vv[0];
#pragma unroll
        for (int ii = 1; ii < 16; ++ii) m = fmaxf(m, vv[ii]);
        float s = 0.0f;
#pragma unroll
        for (int ii = 0; ii < 16; ++ii) s += __expf(vv[ii] - m);
        pm[seg][j] = m;
        ps[seg][j] = s;
        __syncthreads();

        if (seg == 0) {
            float M = fmaxf(fmaxf(pm[0][j], pm[1][j]), fmaxf(pm[2][j], pm[3][j]));
            float S = ps[0][j] * __expf(pm[0][j] - M) + ps[1][j] * __expf(pm[1][j] - M) +
                      ps[2][j] * __expf(pm[2][j] - M) + ps[3][j] * __expf(pm[3][j] - M);
            alpha[j] = M + __logf(S) + emit[(r0 + t) * HK + j];
        }
        __syncthreads();
    }

    if (tid == 0) {
        float M = alpha[0];
        for (int i = 1; i < HK; ++i) M = fmaxf(M, alpha[i]);
        float S = 0.0f;
        for (int i = 0; i < HK; ++i) S += __expf(alpha[i] - M);
        nlse[n] = M + __logf(S);
    }
}

// ---------------------------------------------------------------------------
// Final: out = -mean_n(nlse[n])  (single thread, fixed order)
// ---------------------------------------------------------------------------
__global__ void hmm_final(const float* __restrict__ nlse, float* __restrict__ out) {
    float s = 0.0f;
    for (int i = 0; i < HN; ++i) s += nlse[i];
    out[0] = -s / (float)HN;
}

// ---------------------------------------------------------------------------
extern "C" void kernel_launch(void* const* d_in, const int* in_sizes, int n_in,
                              void* d_out, int out_size, void* d_ws, size_t ws_size,
                              hipStream_t stream) {
    const int*   x        = (const int*)d_in[0];
    const float* emb_w    = (const float*)d_in[1];
    const float* W_trans  = (const float*)d_in[2];
    const float* start_w  = (const float*)d_in[3];
    const float* start_b  = (const float*)d_in[4];
    const float* ecw      = (const float*)d_in[5];
    const float* vocab_w  = (const float*)d_in[6];
    float*       out      = (float*)d_out;

    char* ws = (char*)d_ws;
    size_t off = 0;
    auto take = [&](size_t bytes) {
        size_t cur = off;
        off += (bytes + 255) & ~(size_t)255;
        return cur;
    };
    unsigned short* wfrag = (unsigned short*)(ws + take((size_t)HC * HEP * 2));      // 1 MB
    unsigned short* xeb   = (unsigned short*)(ws + take((size_t)HNT * HEP * 2));     // 4.2 MB
    unsigned short* tran  = (unsigned short*)(ws + take((size_t)HNT * HC * 2));      // 133.7 MB
    float* emit     = (float*)(ws + take((size_t)HNT * HK * 4));                     // 4.2 MB
    float* pmax     = (float*)(ws + take((size_t)125 * HK * 4));
    float* psum     = (float*)(ws + take((size_t)125 * HK * 4));
    float* elog_lse = (float*)(ws + take(256));
    float* alpha0   = (float*)(ws + take(256));
    float* nlse     = (float*)(ws + take(256));
    (void)in_sizes; (void)n_in; (void)out_size; (void)ws_size;

    hmm_prep_w   <<<(HC * HEP) / 256, 256, 0, stream>>>(W_trans, wfrag);
    hmm_prep_xe  <<<(HNT * HEP) / 256, 256, 0, stream>>>(x, emb_w, xeb);
    hmm_gemm_tran<<<HNT / 16, 512, 0, stream>>>(xeb, wfrag, tran);
    hmm_elog_partial<<<HV / 256, 256, 0, stream>>>(ecw, vocab_w, pmax, psum);
    hmm_elog_final  <<<1, HK, 0, stream>>>(pmax, psum, start_w, start_b, elog_lse, alpha0);
    hmm_emit_gather <<<HNT, HK, 0, stream>>>(x, ecw, vocab_w, elog_lse, emit);
    hmm_scan        <<<HN, 256, 0, stream>>>(tran, emit, alpha0, nlse);
    hmm_final       <<<1, 1, 0, stream>>>(nlse, out);
}
